// SelfAttention_38800734552505
// MI455X (gfx1250) — matrix operand order repaired and checked by reading, not compiled, not run on hardware
//
#include <hip/hip_runtime.h>

#define HEADS 16
#define DIM_HEAD 64
#define DIM 1024
#define INNER 1024
#define NSEQ 2048
#define BATCH 4
#define ROWS (BATCH * NSEQ) // 8192

typedef __attribute__((ext_vector_type(16))) __bf16 v16bf;
typedef __attribute__((ext_vector_type(8)))  __bf16 v8bf;
typedef __attribute__((ext_vector_type(2)))  __bf16 v2bf;
typedef __attribute__((ext_vector_type(8)))  float  v8f;
typedef __attribute__((ext_vector_type(4)))  float  f4;
typedef __attribute__((ext_vector_type(4)))  unsigned int u4;

#if defined(__gfx1250__)
#define HAVE_ASYNC_LDS 1
#endif

// 16-byte global(bf16)->LDS copy: ASYNCcnt-tracked DMA on gfx1250
// (GLOBAL_LOAD_ASYNC_TO_LDS_B128, GV mode: vdst = LDS byte address,
//  vaddr = 64-bit global address, saddr = off). VGPR fallback elsewhere.
static __device__ __forceinline__ void copy16_to_lds(__bf16* dst, const __bf16* src) {
#ifdef HAVE_ASYNC_LDS
    unsigned lds_addr = (unsigned)(size_t)dst; // LDS aperture: addr[31:0]
    asm volatile("global_load_async_to_lds_b128 %0, %1, off"
                 :
                 : "v"(lds_addr), "v"(src)
                 : "memory");
#else
    *(u4*)dst = *(const u4*)src;
#endif
}
static __device__ __forceinline__ void copy_wait() {
#ifdef HAVE_ASYNC_LDS
    asm volatile("s_wait_asynccnt 0" ::: "memory");
#endif
}

// pack two f32 -> bf16 pair, single ds_store_b32 (v_cvt_pk_bf16_f32)
static __device__ __forceinline__ void st2(__bf16* p, float a, float b) {
    v2bf t;
    t[0] = (__bf16)a;
    t[1] = (__bf16)b;
    *(v2bf*)p = t;
}

// ---- WMMA fragment loaders from LDS -------------------------------------
// A-matrix 16x32 bf16 (ISA 7.12.2): lane = hh*16+m holds row m,
//   K = {hh*8 .. hh*8+7} in elems 0..7 and {16+hh*8 ..} in elems 8..15.
static __device__ __forceinline__ v16bf frag_a(const __bf16* base, int stride) {
    const int lane = threadIdx.x & 31;
    const int m = lane & 15, hh = lane >> 4;
    const __bf16* p = base + m * stride;
    v16bf f;
    ((v8bf*)&f)[0] = *(const v8bf*)(p + hh * 8);
    ((v8bf*)&f)[1] = *(const v8bf*)(p + 16 + hh * 8);
    return f;
}
// B-matrix 32x16 bf16 stored as [n][k] in LDS: lane = hh*16+n holds col n,
//   K = {hh*8 .. hh*8+7} in elems 0..7 and {16+hh*8 ..} in elems 8..15.
static __device__ __forceinline__ v16bf frag_b(const __bf16* base, int stride) {
    const int lane = threadIdx.x & 31;
    const int n = lane & 15, hh = lane >> 4;
    const __bf16* p = base + n * stride + hh * 8;
    v16bf f;
    ((v8bf*)&f)[0] = *(const v8bf*)(p);
    ((v8bf*)&f)[1] = *(const v8bf*)(p + 16);
    return f;
}

static __device__ __forceinline__ v8f wmma_bf16(v16bf a, v16bf b, v8f c) {
    return __builtin_amdgcn_wmma_f32_16x16x32_bf16(false, a, false, b,
                                                   (short)0, c, false, false);
}

// =========================================================================
// Kernel 1: QKV projection.  C[8192,3072] = x[8192,1024] @ w_qkv[1024,3072]
// 128x64 tile per block (4 waves, 32x64 per wave). Output scattered to
// bf16 qkv[3][B][H][N][64].
// =========================================================================
__global__ __launch_bounds__(128) void qkv_gemm(const float* __restrict__ x,
                                                const float* __restrict__ w,
                                                __bf16* __restrict__ qkv) {
    __shared__ __align__(16) __bf16 As[128 * 32];
    __shared__ __align__(16) __bf16 Bs[64 * 32]; // [n][k]
    const int tid = threadIdx.x;
    const int wave = tid >> 5, lane = tid & 31;
    const int hh = lane >> 4, nl = lane & 15;
    const int rbase = blockIdx.x * 128;
    const int cbase = blockIdx.y * 64;

    v8f acc[2][4] = {};
    for (int kk = 0; kk < DIM; kk += 32) {
#pragma unroll
        for (int i = 0; i < 8; ++i) { // A tile 128x32 f32 -> bf16 (packed st)
            int s = tid + i * 128;
            int r = s >> 3, c = (s & 7) * 4;
            f4 v = *(const f4*)(x + (size_t)(rbase + r) * DIM + kk + c);
            st2(As + r * 32 + c, v.x, v.y);
            st2(As + r * 32 + c + 2, v.z, v.w);
        }
#pragma unroll
        for (int i = 0; i < 4; ++i) { // B tile 32x64 -> transposed [n][k]
            int s = tid + i * 128;
            int k = s >> 4, n4 = (s & 15) * 4;
            f4 v = *(const f4*)(w + (size_t)(kk + k) * (3 * INNER) + cbase + n4);
            Bs[(n4 + 0) * 32 + k] = (__bf16)v.x;
            Bs[(n4 + 1) * 32 + k] = (__bf16)v.y;
            Bs[(n4 + 2) * 32 + k] = (__bf16)v.z;
            Bs[(n4 + 3) * 32 + k] = (__bf16)v.w;
        }
        if (kk + 32 < DIM) { // prefetch next K tiles -> global_prefetch_b8
            __builtin_prefetch(x + (size_t)(rbase + (tid >> 3)) * DIM + kk + 32, 0, 0);
            __builtin_prefetch(w + (size_t)(kk + 32 + (tid >> 4)) * (3 * INNER) + cbase, 0, 0);
        }
        __syncthreads();
        v16bf a0 = frag_a(As + (wave * 32) * 32, 32);
        v16bf a1 = frag_a(As + (wave * 32 + 16) * 32, 32);
#pragma unroll
        for (int t = 0; t < 4; ++t) {
            v16bf b = frag_b(Bs + t * 16 * 32, 32);
            acc[0][t] = wmma_bf16(a0, b, acc[0][t]);
            acc[1][t] = wmma_bf16(a1, b, acc[1][t]);
        }
        __syncthreads();
    }
    // scatter: which/h uniform per block (64-col tiles never straddle)
    const int which = cbase >> 10;
    const int h = (cbase & 1023) >> 6;
#pragma unroll
    for (int sub = 0; sub < 2; ++sub) {
#pragma unroll
        for (int t = 0; t < 4; ++t) {
            int d = t * 16 + nl;
#pragma unroll
            for (int r = 0; r < 8; ++r) {
                int row = rbase + wave * 32 + sub * 16 + hh * 8 + r;
                int b = row >> 11, n = row & (NSEQ - 1);
                size_t idx = ((((size_t)which * BATCH + b) * HEADS + h) * NSEQ + n) * 64 + d;
                qkv[idx] = (__bf16)acc[sub][t][r];
            }
        }
    }
}

// =========================================================================
// Kernel 2: flash attention for one (b,h) and one 64-row query tile.
// =========================================================================
__global__ __launch_bounds__(128) void attention(const __bf16* __restrict__ qkv,
                                                 __bf16* __restrict__ attn) {
    __shared__ __align__(16) __bf16 Qs[64 * 64]; // [m][d]
    __shared__ __align__(16) __bf16 Ks[64 * 64]; // [j][d]
    __shared__ __align__(16) __bf16 Vs[64 * 64]; // [d][j]  (transposed)
    __shared__ __align__(16) __bf16 Ps[64 * 64]; // [m][j]

    const int tid = threadIdx.x;
    const int wave = tid >> 5, lane = tid & 31;
    const int hh = lane >> 4, nl = lane & 15;
    const int bh = blockIdx.y;      // b*H + h, 0..63
    const int q0 = blockIdx.x * 64; // query tile start

    const size_t plane = (size_t)BATCH * HEADS * NSEQ * 64; // 8M elems
    const __bf16* qp = qkv + ((size_t)bh) * NSEQ * 64;
    const __bf16* kp = qkv + plane + ((size_t)bh) * NSEQ * 64;
    const __bf16* vp = qkv + 2 * plane + ((size_t)bh) * NSEQ * 64;

#pragma unroll
    for (int i = 0; i < 4; ++i) { // Q tile: 64x64 bf16, async -> LDS
        int s = tid + i * 128;
        int r = s >> 3, seg = (s & 7) * 8;
        copy16_to_lds(Qs + r * 64 + seg, qp + (size_t)(q0 + r) * 64 + seg);
    }

    float mrow[8], lrow[8];
    v8f oacc[4] = {};
#pragma unroll
    for (int r = 0; r < 8; ++r) { mrow[r] = -1e30f; lrow[r] = 0.f; }

    for (int jt = 0; jt < NSEQ / 64; ++jt) {
#pragma unroll
        for (int i = 0; i < 4; ++i) { // K tile row-major, async -> LDS
            int s = tid + i * 128;
            int r = s >> 3, seg = (s & 7) * 8;
            copy16_to_lds(Ks + r * 64 + seg, kp + (size_t)(jt * 64 + r) * 64 + seg);
        }
#pragma unroll
        for (int i = 0; i < 4; ++i) { // V tile transposed -> [d][j]
            int s = tid + i * 128;
            int j = s >> 3, seg = (s & 7) * 8;
            v8bf v = *(const v8bf*)(vp + (size_t)(jt * 64 + j) * 64 + seg);
#pragma unroll
            for (int e = 0; e < 8; ++e) Vs[(seg + e) * 64 + j] = v[e];
        }
        copy_wait();
        __syncthreads();

        // ---- S = (Q K^T) * scale : 16x64 strip per wave --------------------
        v8f sacc[4] = {};
        v16bf a0 = frag_a(Qs + wave * 16 * 64, 64);
        v16bf a1 = frag_a(Qs + wave * 16 * 64 + 32, 64);
#pragma unroll
        for (int t = 0; t < 4; ++t) {
            v16bf b0 = frag_b(Ks + t * 16 * 64, 64);
            sacc[t] = wmma_bf16(a0, b0, sacc[t]);
            v16bf b1 = frag_b(Ks + t * 16 * 64 + 32, 64);
            sacc[t] = wmma_bf16(a1, b1, sacc[t]);
        }
        const float scale = 0.125f; // 64^-0.5
#pragma unroll
        for (int t = 0; t < 4; ++t)
#pragma unroll
            for (int r = 0; r < 8; ++r) sacc[t][r] *= scale;

        // ---- online softmax (row = wave*16 + hh*8 + r) ---------------------
#pragma unroll
        for (int r = 0; r < 8; ++r) {
            float mx = sacc[0][r];
#pragma unroll
            for (int t = 1; t < 4; ++t) mx = fmaxf(mx, sacc[t][r]);
            mx = fmaxf(mx, __shfl_xor(mx, 1));
            mx = fmaxf(mx, __shfl_xor(mx, 2));
            mx = fmaxf(mx, __shfl_xor(mx, 4));
            mx = fmaxf(mx, __shfl_xor(mx, 8));
            float mnew = fmaxf(mrow[r], mx);
            float corr = __expf(mrow[r] - mnew);
            mrow[r] = mnew;

            float ps = 0.f;
            int prow = (wave * 16 + hh * 8 + r) * 64;
#pragma unroll
            for (int t = 0; t < 4; ++t) {
                float p = __expf(sacc[t][r] - mnew);
                ps += p;
                Ps[prow + t * 16 + nl] = (__bf16)p;
            }
            ps += __shfl_xor(ps, 1);
            ps += __shfl_xor(ps, 2);
            ps += __shfl_xor(ps, 4);
            ps += __shfl_xor(ps, 8);
            lrow[r] = lrow[r] * corr + ps;
#pragma unroll
            for (int t = 0; t < 4; ++t) oacc[t][r] *= corr;
        }

        // ---- O += P @ V (same-wave LDS RAW on Ps: DScnt-ordered) -----------
        v16bf pa0 = frag_a(Ps + wave * 16 * 64, 64);
        v16bf pa1 = frag_a(Ps + wave * 16 * 64 + 32, 64);
#pragma unroll
        for (int t = 0; t < 4; ++t) {
            v16bf vb0 = frag_b(Vs + t * 16 * 64, 64);
            oacc[t] = wmma_bf16(pa0, vb0, oacc[t]);
            v16bf vb1 = frag_b(Vs + t * 16 * 64 + 32, 64);
            oacc[t] = wmma_bf16(pa1, vb1, oacc[t]);
        }
        __syncthreads(); // before next iteration overwrites Ks/Vs
    }

    // ---- epilogue: attn[b*N+n][h*64+d] (bf16) ------------------------------
    const int b = bh >> 4, h = bh & 15;
#pragma unroll
    for (int t = 0; t < 4; ++t) {
        int d = t * 16 + nl;
#pragma unroll
        for (int r = 0; r < 8; ++r) {
            int m = wave * 16 + hh * 8 + r;
            float o = oacc[t][r] / lrow[r];
            attn[((size_t)(b * NSEQ + q0 + m)) * INNER + h * 64 + d] = (__bf16)o;
        }
    }
}

// =========================================================================
// Kernel 3: out = attn[8192,1024](bf16) @ w_out[1024,1024] + b_out  (f32)
// 128x64 tile per block.
// =========================================================================
__global__ __launch_bounds__(128) void out_gemm(const __bf16* __restrict__ attn,
                                                const float* __restrict__ w,
                                                const float* __restrict__ bias,
                                                float* __restrict__ out) {
    __shared__ __align__(16) __bf16 As[128 * 32];
    __shared__ __align__(16) __bf16 Bs[64 * 32]; // [n][k]
    const int tid = threadIdx.x;
    const int wave = tid >> 5, lane = tid & 31;
    const int hh = lane >> 4, nl = lane & 15;
    const int rbase = blockIdx.x * 128;
    const int cbase = blockIdx.y * 64;

    v8f acc[2][4] = {};
    for (int kk = 0; kk < INNER; kk += 32) {
#pragma unroll
        for (int i = 0; i < 4; ++i) { // A tile 128x32 bf16, async -> LDS
            int s = tid + i * 128;
            int r = s >> 2, seg = (s & 3) * 8;
            copy16_to_lds(As + r * 32 + seg,
                          attn + (size_t)(rbase + r) * INNER + kk + seg);
        }
#pragma unroll
        for (int i = 0; i < 4; ++i) { // B tile transpose, f32 -> bf16
            int s = tid + i * 128;
            int k = s >> 4, n4 = (s & 15) * 4;
            f4 v = *(const f4*)(w + (size_t)(kk + k) * DIM + cbase + n4);
            Bs[(n4 + 0) * 32 + k] = (__bf16)v.x;
            Bs[(n4 + 1) * 32 + k] = (__bf16)v.y;
            Bs[(n4 + 2) * 32 + k] = (__bf16)v.z;
            Bs[(n4 + 3) * 32 + k] = (__bf16)v.w;
        }
        if (kk + 32 < INNER) {
            __builtin_prefetch(attn + (size_t)(rbase + (tid >> 2)) * INNER + kk + 32, 0, 0);
            __builtin_prefetch(w + (size_t)(kk + 32 + (tid >> 4)) * DIM + cbase, 0, 0);
        }
        copy_wait();
        __syncthreads();
        v16bf a0 = frag_a(As + (wave * 32) * 32, 32);
        v16bf a1 = frag_a(As + (wave * 32 + 16) * 32, 32);
#pragma unroll
        for (int t = 0; t < 4; ++t) {
            v16bf b = frag_b(Bs + t * 16 * 32, 32);
            acc[0][t] = wmma_bf16(a0, b, acc[0][t]);
            acc[1][t] = wmma_bf16(a1, b, acc[1][t]);
        }
        __syncthreads();
    }
#pragma unroll
    for (int sub = 0; sub < 2; ++sub) {
#pragma unroll
        for (int t = 0; t < 4; ++t) {
            int col = cbase + t * 16 + nl;
            float bb = bias[col];
#pragma unroll
            for (int r = 0; r < 8; ++r) {
                int row = rbase + wave * 32 + sub * 16 + hh * 8 + r;
                out[(size_t)row * DIM + col] = acc[sub][t][r] + bb;
            }
        }
    }
}

// =========================================================================
extern "C" void kernel_launch(void* const* d_in, const int* in_sizes, int n_in,
                              void* d_out, int out_size, void* d_ws, size_t ws_size,
                              hipStream_t stream) {
    const float* x     = (const float*)d_in[0];
    const float* w_qkv = (const float*)d_in[1];
    const float* w_out = (const float*)d_in[2];
    const float* b_out = (const float*)d_in[3];
    float* out = (float*)d_out;

    __bf16* qkv  = (__bf16*)d_ws;                                  // 3*8M bf16 = 48MB
    __bf16* attn = qkv + (size_t)3 * BATCH * HEADS * NSEQ * 64;    // 8M bf16 = 16MB

    qkv_gemm<<<dim3(ROWS / 128, (3 * INNER) / 64), 128, 0, stream>>>(x, w_qkv, qkv);
    attention<<<dim3(NSEQ / 64, BATCH * HEADS), 128, 0, stream>>>(qkv, attn);
    out_gemm<<<dim3(ROWS / 128, DIM / 64), 128, 0, stream>>>(attn, w_out, b_out, out);
}
